// ContrastiveLoss_541165879318
// MI455X (gfx1250) — matrix operand order repaired and checked by reading, not compiled, not run on hardware
//
#include <hip/hip_runtime.h>
#include <hip/hip_bf16.h>
#include <float.h>
#include <stdint.h>

// ---------------------------------------------------------------------------
// ContrastiveLoss on MI455X (gfx1250, wave32, WMMA + TDM)
//   scores = q @ items^T via split-bf16 WMMA (Ah*Bh + Ah*Bl + Al*Bh ~= fp32)
//   A fragments register-resident for the whole N sweep (K=512 -> 256 VGPRs)
//   B tiles pre-converted once, DMA'd by the Tensor Data Mover into a
//   double-buffered LDS region, overlapped with compute.
//   top-2 per row fused into the GEMM (softmax dropped: monotonic).
// ---------------------------------------------------------------------------

typedef __attribute__((ext_vector_type(8)))  __bf16       bf16x8;
typedef __attribute__((ext_vector_type(16))) __bf16       v16bf;
typedef __attribute__((ext_vector_type(8)))  float        v8f;
typedef __attribute__((ext_vector_type(4)))  unsigned int v4u;
typedef __attribute__((ext_vector_type(8)))  int          v8i;
typedef __attribute__((ext_vector_type(4)))  int          v4i;

#define K_DIM            512
#define TILE_N           16
#define WAVES_PER_BLOCK  4
#define ROWS_PER_WAVE    16
#define ROWS_PER_BLOCK   (WAVES_PER_BLOCK * ROWS_PER_WAVE)          // 64
#define THREADS1         (WAVES_PER_BLOCK * 32)                     // 128
#define KCHUNKS          (K_DIM / 32)                               // 16
#define BHALF_BYTES      (TILE_N * K_DIM * 2)                       // 16KB (one hi or lo tile)
#define BBUF_BYTES       (2 * BHALF_BYTES)                          // 32KB (hi+lo)
#define SMEM_BYTES       (WAVES_PER_BLOCK * ROWS_PER_WAVE * K_DIM * 2 * 2) // 128KB

#if defined(__HIP_DEVICE_COMPILE__) && defined(__has_builtin)
#  if __has_builtin(__builtin_amdgcn_tensor_load_to_lds)
#    define USE_TDM 1
#  endif
#endif
#ifndef USE_TDM
#  define USE_TDM 0
#endif

__device__ __forceinline__ void bf16_split(float f, __bf16& h, __bf16& l) {
  h = (__bf16)f;                 // RNE (hi)
  l = (__bf16)(f - (float)h);    // residual (lo)
}

#if USE_TDM
// 2D TDM descriptor: tile = TILE_N rows x K_DIM bf16, row stride K_DIM, into LDS.
__device__ __forceinline__ void tdm_load_2d(unsigned ldsOff, const __bf16* src,
                                            int row0, int nRowsTensor) {
  unsigned long long ga = (unsigned long long)(uintptr_t)(src + (size_t)row0 * K_DIM);
  v4u g0 = { 1u,                                           // count=1, no gather
             ldsOff,                                       // lds_addr (bytes)
             (unsigned)(ga & 0xFFFFFFFFull),               // global_addr[31:0]
             (unsigned)((ga >> 32) & 0x1FFFFFFull) | (2u << 30) }; // addr[56:32] | type=2
  v8i g1 = { (int)0x10000,                                 // data_size=1 (2 bytes)
             (int)((unsigned)(K_DIM & 0xFFFF) << 16),      // tensor_dim0[15:0] @ [31:16]
             (int)(((unsigned)nRowsTensor & 0xFFFF) << 16),// dim0[31:16]=0 | tensor_dim1[15:0]
             (int)((unsigned)(K_DIM & 0xFFFF) << 16),      // dim1 hi=0 | tile_dim0 @ [31:16]
             TILE_N,                                       // tile_dim1 (tile_dim2=0)
             K_DIM,                                        // tensor_dim0_stride[31:0]
             0, 0 };                                       // stride hi / dim1_stride
  v4i z = { 0, 0, 0, 0 };
#if __clang_major__ >= 23
  v8i z8 = { 0, 0, 0, 0, 0, 0, 0, 0 };
  __builtin_amdgcn_tensor_load_to_lds(g0, g1, z, z, z8, 0);
#else
  __builtin_amdgcn_tensor_load_to_lds(g0, g1, z, z, 0);
#endif
}
#endif

// ---------------------------------------------------------------------------
// Pre-pass: items fp32 -> bf16 hi/lo (done once, not once per block).
// ---------------------------------------------------------------------------
__global__ __launch_bounds__(256)
void convert_items_kernel(const float* __restrict__ items,
                          __bf16* __restrict__ hi, __bf16* __restrict__ lo, int n)
{
  int i = blockIdx.x * 256 + threadIdx.x;
  if (i < n) {
    __bf16 h, l; bf16_split(items[i], h, l);
    hi[i] = h; lo[i] = l;
  }
}

// ---------------------------------------------------------------------------
// Kernel 1: fused scores + running top-2.
// Block = 4 waves, wave owns 16 query rows. Phase 1 stages q as bf16 hi/lo in
// LDS then pulls the full-K A fragments into registers. Phase 2 reuses the
// same LDS as a double-buffered B tile fed by the TDM.
// ---------------------------------------------------------------------------
__global__ __launch_bounds__(THREADS1, 2)
void top2_scores_kernel(const float* __restrict__ q,
                        const __bf16* __restrict__ itemsHi,
                        const __bf16* __restrict__ itemsLo,
                        int2* __restrict__ top2, int nItems)
{
  extern __shared__ __align__(16) char smem[];
  const int tid    = threadIdx.x;
  const int w      = tid >> 5;
  const int lane   = tid & 31;
  const int halfHi = (lane >= 16);
  const int col    = lane & 15;
  const int rowBase = blockIdx.x * ROWS_PER_BLOCK + w * ROWS_PER_WAVE;

  // ---- Phase 1: stage q rows (coalesced fp32 load, bf16 hi/lo split) -------
  __bf16* aHi = (__bf16*)smem;                                   // [4][16][512]
  __bf16* aLo = aHi + WAVES_PER_BLOCK * ROWS_PER_WAVE * K_DIM;
  for (int i = lane; i < ROWS_PER_WAVE * K_DIM; i += 32) {
    int r = i >> 9, k = i & (K_DIM - 1);
    float f = q[(size_t)(rowBase + r) * K_DIM + k];
    __bf16 h, l; bf16_split(f, h, l);
    aHi[w * ROWS_PER_WAVE * K_DIM + i] = h;
    aLo[w * ROWS_PER_WAVE * K_DIM + i] = l;
  }

  // A fragment layout (wave32): lane = M row (mod 16); lanes 0-15 take K
  // 0-7/16-23, lanes 16-31 take K 8-15/24-31 of each 32-wide chunk.
  const __bf16* aRowHi = aHi + (size_t)(w * ROWS_PER_WAVE + col) * K_DIM;
  const __bf16* aRowLo = aLo + (size_t)(w * ROWS_PER_WAVE + col) * K_DIM;
  const int aOff = halfHi ? 8  : 0;
  const int bOff = halfHi ? 8  : 0;

  v16bf Ah[KCHUNKS], Al[KCHUNKS];          // 256 VGPRs, live for whole N sweep
#pragma unroll
  for (int kc = 0; kc < KCHUNKS; ++kc) {
    const int kb = kc * 32;
    bf16x8 h0 = *(const bf16x8*)(aRowHi + kb + aOff);
    bf16x8 h1 = *(const bf16x8*)(aRowHi + kb + aOff + 16);
    bf16x8 l0 = *(const bf16x8*)(aRowLo + kb + aOff);
    bf16x8 l1 = *(const bf16x8*)(aRowLo + kb + aOff + 16);
    Ah[kc] = __builtin_shufflevector(h0, h1, 0,1,2,3,4,5,6,7,8,9,10,11,12,13,14,15);
    Al[kc] = __builtin_shufflevector(l0, l1, 0,1,2,3,4,5,6,7,8,9,10,11,12,13,14,15);
  }
  __syncthreads();   // everyone done with phase-1 LDS -> reuse as B buffers

  float b1[8], b2[8]; int i1[8], i2[8];
#pragma unroll
  for (int r = 0; r < 8; ++r) { b1[r] = -FLT_MAX; b2[r] = -FLT_MAX; i1[r] = 0; i2[r] = 1; }

  const int nSteps = nItems / TILE_N;

  // ---- preload tile 0 into buffer 0 ----------------------------------------
#if USE_TDM
  if (w == 0) {
    tdm_load_2d(0u,                    itemsHi, 0, nItems);
    tdm_load_2d((unsigned)BHALF_BYTES, itemsLo, 0, nItems);
  }
#else
  for (int i = tid; i < TILE_N * K_DIM / 8; i += THREADS1) {
    ((bf16x8*)smem)[i]                         = ((const bf16x8*)itemsHi)[i];
    ((bf16x8*)(smem + BHALF_BYTES))[i]         = ((const bf16x8*)itemsLo)[i];
  }
#endif

  for (int step = 0; step < nSteps; ++step) {
    const int cur = step & 1, nxt = cur ^ 1;
    const unsigned curOff = (unsigned)(cur * BBUF_BYTES);
    const unsigned nxtOff = (unsigned)(nxt * BBUF_BYTES);

    // Issue next tile's DMA, then wait for current tile (overlap with compute
    // happens because the DMA for step+1 runs while we WMMA on step).
#if USE_TDM
    if (w == 0) {
      if (step + 1 < nSteps) {
        tdm_load_2d(nxtOff,               itemsHi, (step + 1) * TILE_N, nItems);
        tdm_load_2d(nxtOff + BHALF_BYTES, itemsLo, (step + 1) * TILE_N, nItems);
        __builtin_amdgcn_s_wait_tensorcnt(2);   // current pair done, next pair in flight
      } else {
        __builtin_amdgcn_s_wait_tensorcnt(0);
      }
    }
#else
    if (step + 1 < nSteps) {
      const __bf16* sh = itemsHi + (size_t)(step + 1) * TILE_N * K_DIM;
      const __bf16* sl = itemsLo + (size_t)(step + 1) * TILE_N * K_DIM;
      for (int i = tid; i < TILE_N * K_DIM / 8; i += THREADS1) {
        ((bf16x8*)(smem + nxtOff))[i]               = ((const bf16x8*)sh)[i];
        ((bf16x8*)(smem + nxtOff + BHALF_BYTES))[i] = ((const bf16x8*)sl)[i];
      }
    }
#endif
    __syncthreads();   // current tile visible to all waves

    // B fragment layout: lane = N col (mod 16); lanes 0-15 take K 0-7/16-23,
    // lanes 16-31 take K 8-15/24-31 of each chunk (same order as A).
    const __bf16* bColHi = (const __bf16*)(smem + curOff)               + (size_t)col * K_DIM;
    const __bf16* bColLo = (const __bf16*)(smem + curOff + BHALF_BYTES) + (size_t)col * K_DIM;

    v8f c = {0.f, 0.f, 0.f, 0.f, 0.f, 0.f, 0.f, 0.f};
#pragma unroll
    for (int kc = 0; kc < KCHUNKS; ++kc) {
      const int kb = kc * 32;
      bf16x8 bh0 = *(const bf16x8*)(bColHi + kb + bOff);
      bf16x8 bh1 = *(const bf16x8*)(bColHi + kb + bOff + 16);
      bf16x8 bl0 = *(const bf16x8*)(bColLo + kb + bOff);
      bf16x8 bl1 = *(const bf16x8*)(bColLo + kb + bOff + 16);
      v16bf Bh = __builtin_shufflevector(bh0, bh1, 0,1,2,3,4,5,6,7,8,9,10,11,12,13,14,15);
      v16bf Bl = __builtin_shufflevector(bl0, bl1, 0,1,2,3,4,5,6,7,8,9,10,11,12,13,14,15);
      c = __builtin_amdgcn_wmma_f32_16x16x32_bf16(false, Ah[kc], false, Bh, (short)0, c, false, false);
      c = __builtin_amdgcn_wmma_f32_16x16x32_bf16(false, Ah[kc], false, Bl, (short)0, c, false, false);
      c = __builtin_amdgcn_wmma_f32_16x16x32_bf16(false, Al[kc], false, Bh, (short)0, c, false, false);
    }

    // Branchless running top-2 (row r+8*halfHi, column n).
    const int n = step * TILE_N + col;
#pragma unroll
    for (int r = 0; r < 8; ++r) {
      float v  = c[r];
      bool  t1 = v > b1[r];
      bool  t2 = v > b2[r];
      float nb2 = t1 ? b1[r] : (t2 ? v : b2[r]);
      int   ni2 = t1 ? i1[r] : (t2 ? n : i2[r]);
      b1[r] = t1 ? v : b1[r];
      i1[r] = t1 ? n : i1[r];
      b2[r] = nb2;
      i2[r] = ni2;
    }
    __syncthreads();   // all waves done with current buffer before it is reused
  }

  // Merge top-2 across the 16 lanes of each half (xor masks stay in-half).
#pragma unroll
  for (int m = 1; m < 16; m <<= 1) {
#pragma unroll
    for (int r = 0; r < 8; ++r) {
      float ov1 = __shfl_xor(b1[r], m, 32);
      int   oi1 = __shfl_xor(i1[r], m, 32);
      float ov2 = __shfl_xor(b2[r], m, 32);
      int   oi2 = __shfl_xor(i2[r], m, 32);
      bool otherTop = (ov1 > b1[r]) || (ov1 == b1[r] && oi1 < i1[r]);
      bool mineSecond = (b1[r] > ov2) || (b1[r] == ov2 && i1[r] < oi2);
      float tb2 = otherTop ? (mineSecond ? b1[r] : ov2)
                           : (((ov1 > b2[r]) || (ov1 == b2[r] && oi1 < i2[r])) ? ov1 : b2[r]);
      int   ti2 = otherTop ? (mineSecond ? i1[r] : oi2)
                           : (((ov1 > b2[r]) || (ov1 == b2[r] && oi1 < i2[r])) ? oi1 : i2[r]);
      float tb1 = otherTop ? ov1 : b1[r];
      int   ti1 = otherTop ? oi1 : i1[r];
      b1[r] = tb1; i1[r] = ti1; b2[r] = tb2; i2[r] = ti2;
    }
  }

  if (col == 0) {
    const int rbase = rowBase + (halfHi ? 8 : 0);
#pragma unroll
    for (int r = 0; r < 8; ++r) top2[rbase + r] = make_int2(i1[r], i2[r]);
  }
}

// ---------------------------------------------------------------------------
// Kernel 2: per-row hinge loss; one wave per query row; items hit in L2.
// ---------------------------------------------------------------------------
__global__ __launch_bounds__(256)
void hinge_loss_kernel(const float* __restrict__ q,
                       const float* __restrict__ items,
                       const int2* __restrict__ top2,
                       float* __restrict__ out, int M)
{
  const int row  = blockIdx.x * 8 + (threadIdx.x >> 5);
  const int lane = threadIdx.x & 31;
  if (row >= M) return;

  const float* qr = q + (size_t)row * K_DIM;
  const int2  t   = top2[row];
  const float* p  = items + (size_t)t.x * K_DIM;
  const float* ng = items + (size_t)t.y * K_DIM;

  float sp = 0.f, sn = 0.f;
  for (int k = lane; k < K_DIM; k += 32) {
    float qv = qr[k];
    float dp = qv - p[k]  + 1e-6f;
    float dn = qv - ng[k] + 1e-6f;
    sp = fmaf(dp, dp, sp);
    sn = fmaf(dn, dn, sn);
  }
#pragma unroll
  for (int m = 16; m; m >>= 1) {
    sp += __shfl_xor(sp, m, 32);
    sn += __shfl_xor(sn, m, 32);
  }
  if (lane == 0) {
    float loss = fmaxf(sqrtf(sp) - sqrtf(sn) + 1.0f, 0.0f);   // MARGIN = 1.0
    atomicAdd(out, loss * (1.0f / (float)M));
  }
}

__global__ void zero_out_kernel(float* o) { if (threadIdx.x == 0) o[0] = 0.0f; }

// ---------------------------------------------------------------------------
extern "C" void kernel_launch(void* const* d_in, const int* in_sizes, int n_in,
                              void* d_out, int out_size, void* d_ws, size_t ws_size,
                              hipStream_t stream)
{
  (void)n_in; (void)out_size; (void)ws_size;
  const float* q     = (const float*)d_in[0];   // (32768, 512) fp32
  const float* items = (const float*)d_in[1];   // (4096, 512)  fp32
  float*       out   = (float*)d_out;

  const int M      = in_sizes[0] / K_DIM;       // 32768
  const int nItems = in_sizes[1] / K_DIM;       // 4096

  // ws layout: itemsHi (4MB) | itemsLo (4MB) | top2 (256KB)
  __bf16* itemsHi = (__bf16*)d_ws;
  __bf16* itemsLo = itemsHi + (size_t)nItems * K_DIM;
  int2*   top2    = (int2*)(itemsLo + (size_t)nItems * K_DIM);

  zero_out_kernel<<<1, 32, 0, stream>>>(out);

  const int nconv = nItems * K_DIM;
  convert_items_kernel<<<(nconv + 255) / 256, 256, 0, stream>>>(items, itemsHi, itemsLo, nconv);

  top2_scores_kernel<<<M / ROWS_PER_BLOCK, THREADS1, SMEM_BYTES, stream>>>(
      q, itemsHi, itemsLo, top2, nItems);

  hinge_loss_kernel<<<M / 8, 256, 0, stream>>>(q, items, top2, out, M);
}
